// AE_gnnrnn_6811818132039
// MI455X (gfx1250) — matrix operand order repaired and checked by reading, not compiled, not run on hardware
//
#include <hip/hip_runtime.h>
#include <hip/hip_bf16.h>
#include <cstdint>

// ---------------------------------------------------------------- types
typedef unsigned short u16;
typedef __attribute__((ext_vector_type(16))) __bf16        v16bf;
typedef __attribute__((ext_vector_type(8)))  float         v8f;
typedef __attribute__((ext_vector_type(8)))  unsigned short us8;

#define BB   128
#define TT   256
#define EMBD 128
#define HHD  128
#define G4   512
#define NMX  4096
#define EEDG 1024

// ---------------------------------------------------------------- helpers
__device__ __forceinline__ u16 f2bf(float f) {
  unsigned u = __float_as_uint(f);
  u += 0x7FFFu + ((u >> 16) & 1u);          // round-to-nearest-even
  return (u16)(u >> 16);
}
__device__ __forceinline__ float sigm(float x) { return 1.f / (1.f + __expf(-x)); }

union BFrag { us8 h[2]; v16bf v; };

// A-matrix 16x32 bf16 fragment (row-major A, lda = K)
__device__ __forceinline__ v16bf loadAfrag(const u16* A, int lda, int row, int kb, int lane) {
  int half = lane >> 4;
  const u16* p = A + (size_t)row * lda + kb + half * 8;
  BFrag f; f.h[0] = *(const us8*)p; f.h[1] = *(const us8*)(p + 16);
  return f.v;
}
// B-matrix 32x16 bf16 fragment from Bt (N,K) row-major (i.e. W as (out,in))
__device__ __forceinline__ v16bf loadBfrag(const u16* Bt, int ldb, int col, int kb, int lane) {
  int half = lane >> 4;
  const u16* p = Bt + (size_t)col * ldb + kb + half * 8;
  BFrag f; f.h[0] = *(const us8*)p; f.h[1] = *(const us8*)(p + 16);
  return f.v;
}

// ---------------------------------------------------------------- simple kernels
__global__ void k_f2bf(const float* __restrict__ s, u16* __restrict__ d, size_t n) {
  size_t i = (size_t)blockIdx.x * blockDim.x + threadIdx.x;
  if (i < n) d[i] = f2bf(s[i]);
}

__global__ void k_zero(float* p, size_t n) {
  size_t i = (size_t)blockIdx.x * blockDim.x + threadIdx.x;
  if (i < n) p[i] = 0.f;
}

// x_bf16[b,t,:] = bf16(emb[seq[b,t],:])
__global__ void k_embed_bf(const int* __restrict__ seq, const float* __restrict__ emb,
                           u16* __restrict__ x, size_t n) {
  size_t i = (size_t)blockIdx.x * blockDim.x + threadIdx.x;
  if (i >= n) return;
  int c = (int)(i & (EMBD - 1));
  size_t bt = i >> 7;
  x[i] = f2bf(emb[(size_t)seq[bt] * EMBD + c]);
}

// dst[b,t,dstOff+c] = src[b, ridx(b,t), srcOff+c]; ridx = clip(len-1-t) or T-1-t
__global__ void k_rev_gather_bf(const u16* __restrict__ src, u16* __restrict__ dst,
                                const int* __restrict__ lengths, size_t n, int C,
                                int srcStride, int srcOff, int dstStride, int dstOff, int T) {
  size_t i = (size_t)blockIdx.x * blockDim.x + threadIdx.x;
  if (i >= n) return;
  int c = (int)(i % C);
  size_t bt = i / C;
  int t = (int)(bt % T);
  int b = (int)(bt / T);
  int r;
  if (lengths) { r = lengths[b] - 1 - t; r = r < 0 ? 0 : (r > T - 1 ? T - 1 : r); }
  else         { r = T - 1 - t; }
  dst[((size_t)b * T + t) * dstStride + dstOff + c] =
      src[((size_t)b * T + r) * srcStride + srcOff + c];
}

// decoder layer-0 input projection: xp[b,t,g] = seq[b, t or T-1-t] * Wih[g] + bih[g] + bhh[g]
__global__ void k_dec_xp(const int* __restrict__ seq, const float* __restrict__ Wih,
                         const float* __restrict__ bih, const float* __restrict__ bhh,
                         float* __restrict__ xp, int rev, size_t n, int T) {
  size_t i = (size_t)blockIdx.x * blockDim.x + threadIdx.x;
  if (i >= n) return;
  int g = (int)(i & (G4 - 1));
  size_t bt = i >> 9;
  int t = (int)(bt % T);
  int b = (int)(bt / T);
  int tt = rev ? (T - 1 - t) : t;
  float s = (float)seq[(size_t)b * T + tt];
  xp[i] = s * Wih[g] + bih[g] + bhh[g];
}

// ---------------------------------------------------------------- WMMA GEMM
// C(M,N) f32 = A(M,K)bf16 @ Bt(N,K)bf16^T + bias0[n] + bias1[n].  K % 64 == 0.
// wave handles 16 rows x 64 cols; two-phase ping-pong over K (no register
// rotation copies): phase-1 fragments load before phase-0 WMMAs and vice versa.
__global__ __launch_bounds__(256)
void k_wmma_gemm(const u16* __restrict__ A, const u16* __restrict__ Bt,
                 const float* __restrict__ bias0, const float* __restrict__ bias1,
                 float* __restrict__ C, int M, int N, int K) {
  int wave = blockIdx.x * (blockDim.x >> 5) + (threadIdx.x >> 5);
  int lane = threadIdx.x & 31;
  int nq = N >> 6;
  int tot = (M >> 4) * nq;
  if (wave >= tot) return;                       // wave-uniform exit
  int wm = wave / nq, wn = wave % nq;
  int row = wm * 16 + (lane & 15);
  int colBase = wn * 64 + (lane & 15);
  v8f acc[4];
  for (int i = 0; i < 4; i++) acc[i] = 0.f;

  v16bf a0 = loadAfrag(A, K, row, 0, lane);
  v16bf b0[4];
  for (int i = 0; i < 4; i++) b0[i] = loadBfrag(Bt, K, colBase + i * 16, 0, lane);

  for (int kb = 0; kb < K - 64; kb += 64) {
    v16bf a1 = loadAfrag(A, K, row, kb + 32, lane);
    v16bf b1[4];
    for (int i = 0; i < 4; i++) b1[i] = loadBfrag(Bt, K, colBase + i * 16, kb + 32, lane);
    for (int i = 0; i < 4; i++)
      acc[i] = __builtin_amdgcn_wmma_f32_16x16x32_bf16(
          false, a0, false, b0[i], (short)0, acc[i], false, false);
    a0 = loadAfrag(A, K, row, kb + 64, lane);    // refill phase-0 regs (no copies)
    for (int i = 0; i < 4; i++) b0[i] = loadBfrag(Bt, K, colBase + i * 16, kb + 64, lane);
    for (int i = 0; i < 4; i++)
      acc[i] = __builtin_amdgcn_wmma_f32_16x16x32_bf16(
          false, a1, false, b1[i], (short)0, acc[i], false, false);
  }
  {                                              // final K-64 pair
    v16bf a1 = loadAfrag(A, K, row, K - 32, lane);
    v16bf b1[4];
    for (int i = 0; i < 4; i++) b1[i] = loadBfrag(Bt, K, colBase + i * 16, K - 32, lane);
    for (int i = 0; i < 4; i++)
      acc[i] = __builtin_amdgcn_wmma_f32_16x16x32_bf16(
          false, a0, false, b0[i], (short)0, acc[i], false, false);
    for (int i = 0; i < 4; i++)
      acc[i] = __builtin_amdgcn_wmma_f32_16x16x32_bf16(
          false, a1, false, b1[i], (short)0, acc[i], false, false);
  }

  int rb = wm * 16 + ((lane >> 4) << 3);
  for (int i = 0; i < 4; i++) {
    int col = wn * 64 + i * 16 + (lane & 15);
    float badd = (bias0 ? bias0[col] : 0.f) + (bias1 ? bias1[col] : 0.f);
    for (int j = 0; j < 8; j++)
      C[(size_t)(rb + j) * N + col] = acc[i][j] + badd;
  }
}

// ---------------------------------------------------------------- LSTM scan
// One workgroup (32 waves) runs the whole T-step recurrence for one direction.
// gates(128x512) = h(128x128) @ Whh(512x128)^T via WMMA.
// Whh (128 KB bf16) is staged once into LDS (CDNA5 WGP has 320 KB), h is staged
// bf16 in LDS each step -> the serial inner loop is pure ds_load + v_wmma.
// wave w owns mTile = w>>2 and nTiles {w&3, (w&3)+4} of the 8x8 H-tile grid;
// for each owned (m,n) it computes the 4 gate tiles -> gate math is lane-local.
__global__ __launch_bounds__(1024)
void k_lstm_scan(const float* __restrict__ xp,          // (B,T,512) f32 (biases folded)
                 const u16* __restrict__ Whh,           // (512,128) bf16
                 const int* __restrict__ lengths,       // null => mask 1
                 const float* __restrict__ h0, const float* __restrict__ c0, int sStride,
                 u16* __restrict__ outBf, int outStride, int outOff,
                 float* __restrict__ hT, float* __restrict__ cT, int tStride, int T) {
  __shared__ u16 hbf[HHD * HHD];                         // 32 KB
  __shared__ u16 wsh[G4 * HHD];                          // 128 KB (Whh resident)
  const int tid = threadIdx.x, wave = tid >> 5, lane = tid & 31;
  const int mTile = wave >> 2, nQuad = wave & 3;
  const int lrow = lane & 15, lhalf = lane >> 4;
  const int rbase = mTile * 16 + lhalf * 8;              // + j = batch index
  float hReg[2][8], cReg[2][8];

  // stage Whh into LDS (8192 us8 vectors / 1024 threads = 8 each)
  {
    const us8* src = (const us8*)Whh;
    us8* dst = (us8*)wsh;
    for (int i = tid; i < (G4 * HHD) / 8; i += 1024) dst[i] = src[i];
  }

  for (int p = 0; p < 2; p++) {
    int nl = (nQuad + p * 4) * 16 + lrow;
    for (int j = 0; j < 8; j++) {
      int b = rbase + j;
      float h = h0 ? h0[(size_t)b * sStride + nl] : 0.f;
      float c = c0 ? c0[(size_t)b * sStride + nl] : 0.f;
      hReg[p][j] = h; cReg[p][j] = c;
      hbf[b * HHD + nl] = f2bf(h);
    }
  }
  __syncthreads();

  const int aRow = mTile * 16 + lrow;
  for (int t = 0; t < T; t++) {
    // A fragments of h for this wave's 16 rows, K = 0..127
    v16bf afr[4];
    for (int ks = 0; ks < 4; ks++) {
      const u16* p = &hbf[aRow * HHD + ks * 32 + lhalf * 8];
      BFrag f; f.h[0] = *(const us8*)p; f.h[1] = *(const us8*)(p + 16);
      afr[ks] = f.v;
    }
    __syncthreads();                                     // reads done before writes

    for (int p = 0; p < 2; p++) {
      int nl = (nQuad + p * 4) * 16 + lrow;
      v8f acc[4];
      for (int g = 0; g < 4; g++) acc[g] = 0.f;
      for (int g = 0; g < 4; g++) {
        const u16* wrow = &wsh[(size_t)(g * HHD + nl) * HHD];
        for (int ks = 0; ks < 4; ks++) {
          const u16* q = wrow + ks * 32 + lhalf * 8;
          BFrag f; f.h[0] = *(const us8*)q; f.h[1] = *(const us8*)(q + 16);
          acc[g] = __builtin_amdgcn_wmma_f32_16x16x32_bf16(
              false, afr[ks], false, f.v, (short)0, acc[g], false, false);
        }
      }
      for (int j = 0; j < 8; j++) {
        int b = rbase + j;
        const float* xr = xp + ((size_t)b * T + t) * G4;
        float i_ = sigm(acc[0][j] + xr[nl]);
        float f_ = sigm(acc[1][j] + xr[128 + nl]);
        float g_ = tanhf(acc[2][j] + xr[256 + nl]);
        float o_ = sigm(acc[3][j] + xr[384 + nl]);
        float cn = f_ * cReg[p][j] + i_ * g_;
        float hn = o_ * tanhf(cn);
        float m  = lengths ? ((t < lengths[b]) ? 1.f : 0.f) : 1.f;
        float h2 = m * hn + (1.f - m) * hReg[p][j];
        float c2 = m * cn + (1.f - m) * cReg[p][j];
        hReg[p][j] = h2; cReg[p][j] = c2;
        u16 hb = f2bf(h2);
        hbf[b * HHD + nl] = hb;
        if (outBf) outBf[((size_t)b * T + t) * outStride + outOff + nl] = hb;
      }
    }
    __syncthreads();                                     // writes done before next reads
  }

  if (hT) {
    for (int p = 0; p < 2; p++) {
      int nl = (nQuad + p * 4) * 16 + lrow;
      for (int j = 0; j < 8; j++) {
        int b = rbase + j;
        hT[(size_t)b * tStride + nl] = hReg[p][j];
        cT[(size_t)b * tStride + nl] = cReg[p][j];
      }
    }
  }
}

// ---------------------------------------------------------------- GNN (scalar)
__global__ void k_gemm_naive(const float* __restrict__ X, const float* __restrict__ W,
                             const float* __restrict__ bias, float* __restrict__ Y,
                             int M, int N, int K) {
  int i = blockIdx.x * blockDim.x + threadIdx.x;
  if (i >= M * N) return;
  int m = i / N, n = i % N;
  float s = bias ? bias[n] : 0.f;
  for (int k = 0; k < K; k++) s += X[(size_t)m * K + k] * W[(size_t)n * K + k];
  Y[i] = s;
}

__global__ void k_deg(const int* __restrict__ ei, float* __restrict__ deg, int E, int B) {
  int i = blockIdx.x * blockDim.x + threadIdx.x;
  if (i >= E + B) return;
  int col = (i < E) ? ei[E + i] : (i - E);               // self-loops appended
  atomicAdd(&deg[col], 1.f);
}

__global__ void k_dinv(float* deg, int B) {
  int i = blockIdx.x * blockDim.x + threadIdx.x;
  if (i >= B) return;
  float d = deg[i];
  deg[i] = d > 0.f ? rsqrtf(d) : 0.f;
}

__global__ void k_scatter(const int* __restrict__ ei, const float* __restrict__ dinv,
                          const float* __restrict__ xw, float* __restrict__ out,
                          int E, int B, int C) {
  int i = blockIdx.x * blockDim.x + threadIdx.x;
  if (i >= (E + B) * C) return;
  int e = i / C, c = i % C;
  int r  = (e < E) ? ei[e]     : (e - E);
  int cl = (e < E) ? ei[E + e] : (e - E);
  atomicAdd(&out[(size_t)cl * C + c], xw[(size_t)r * C + c] * dinv[r] * dinv[cl]);
}

__global__ void k_bias_lrelu(float* __restrict__ y, const float* __restrict__ b, int M, int N) {
  int i = blockIdx.x * blockDim.x + threadIdx.x;
  if (i >= M * N) return;
  float v = y[i] + b[i % N];
  y[i] = v > 0.f ? v : 0.01f * v;
}

// ---------------------------------------------------------------- host
extern "C" void kernel_launch(void* const* d_in, const int* in_sizes, int n_in,
                              void* d_out, int out_size, void* d_ws, size_t ws_size,
                              hipStream_t stream) {
  (void)in_sizes; (void)n_in; (void)out_size; (void)ws_size;
  const int T = TT, Bn = BB, E = EEDG;
  const size_t BT = (size_t)Bn * T;

  // ---- unpack inputs (jax tree flatten order of setup_inputs dict)
  int ii = 0;
  const int*   seq  = (const int*)d_in[ii++];
  const int*   len  = (const int*)d_in[ii++];
  const int*   eidx = (const int*)d_in[ii++];
  const float* emb  = (const float*)d_in[ii++];
  const float* eW[2][2][4];
  for (int l = 0; l < 2; l++) for (int d = 0; d < 2; d++) for (int k = 0; k < 4; k++)
    eW[l][d][k] = (const float*)d_in[ii++];
  const float* g1p[6]; for (int k = 0; k < 6; k++) g1p[k] = (const float*)d_in[ii++];
  const float* g2p[6]; for (int k = 0; k < 6; k++) g2p[k] = (const float*)d_in[ii++];
  const float* p1W = (const float*)d_in[ii++];
  const float* p1b = (const float*)d_in[ii++];
  const float* p2W = (const float*)d_in[ii++];
  const float* p2b = (const float*)d_in[ii++];
  const float* dW[2][2][4];
  for (int l = 0; l < 2; l++) for (int d = 0; d < 2; d++) for (int k = 0; k < 4; k++)
    dW[l][d][k] = (const float*)d_in[ii++];
  const float* prW = (const float*)d_in[ii++];
  const float* prb = (const float*)d_in[ii++];
  float* out = (float*)d_out;

  // ---- workspace carve
  char* wp = (char*)d_ws;
  auto take = [&](size_t bytes) -> void* {
    void* p = (void*)wp; wp += (bytes + 255) & ~(size_t)255; return p;
  };
  u16*   xbf   = (u16*)  take(BT * EMBD * 2);
  u16*   xrev  = (u16*)  take(BT * 256 * 2);
  float* xpF   = (float*)take(BT * G4 * 4);          // shared fwd/bwd (serial use)
  u16*   cat0  = (u16*)  take(BT * 256 * 2);
  u16*   cat1  = (u16*)  take(BT * 256 * 2);
  u16*   tmpO  = (u16*)  take(BT * HHD * 2);
  float* hncat = (float*)take((size_t)Bn * 512 * 4);
  float* cncat = (float*)take((size_t)Bn * 512 * 4);
  float* projH = (float*)take((size_t)Bn * 512 * 4);
  float* projC = (float*)take((size_t)Bn * 512 * 4);
  float* deg   = (float*)take((size_t)Bn * 4);
  float* gxw   = (float*)take((size_t)Bn * 32 * 4);
  float* gh1   = (float*)take((size_t)Bn * 16 * 4);
  float* gh2   = (float*)take((size_t)Bn * 32 * 4);
  float* goutH = (float*)take((size_t)Bn * 512 * 4);
  float* goutC = (float*)take((size_t)Bn * 512 * 4);
  u16*   gbf   = (u16*)  take((size_t)Bn * 512 * 2);
  u16* eWihBf[2][2], *eWhhBf[2][2], *dWhhBf[2][2], *dWihBf[2];
  for (int l = 0; l < 2; l++) for (int d = 0; d < 2; d++) {
    eWihBf[l][d] = (u16*)take((size_t)G4 * (l ? 256 : EMBD) * 2);
    eWhhBf[l][d] = (u16*)take((size_t)G4 * HHD * 2);
    dWhhBf[l][d] = (u16*)take((size_t)G4 * HHD * 2);
  }
  for (int d = 0; d < 2; d++) dWihBf[d] = (u16*)take((size_t)G4 * 256 * 2);
  u16* p1Wbf = (u16*)take((size_t)512 * 512 * 2);
  u16* p2Wbf = (u16*)take((size_t)512 * 512 * 2);
  u16* prWbf = (u16*)take((size_t)NMX * 256 * 2);

  auto NB = [](size_t n, int t) { return (unsigned)((n + (size_t)t - 1) / (size_t)t); };
  auto cvt = [&](const float* s, u16* d, size_t n) {
    k_f2bf<<<NB(n, 256), 256, 0, stream>>>(s, d, n);
  };
  auto gemm = [&](const u16* A, const u16* Bt, const float* b0, const float* b1,
                  float* C, int M, int N, int K) {
    unsigned waves = (unsigned)((M >> 4) * (N >> 6));
    k_wmma_gemm<<<(waves + 7) / 8, 256, 0, stream>>>(A, Bt, b0, b1, C, M, N, K);
  };
  auto scan = [&](const float* xp, const u16* whh, const int* ln, const float* h0,
                  const float* c0, int sStr, u16* ob, int oStr, int oOff,
                  float* hT, float* cT) {
    k_lstm_scan<<<1, 1024, 0, stream>>>(xp, whh, ln, h0, c0, sStr,
                                        ob, oStr, oOff, hT, cT, 512, T);
  };

  // ---- weight conversions to bf16
  for (int l = 0; l < 2; l++) for (int d = 0; d < 2; d++) {
    cvt(eW[l][d][0], eWihBf[l][d], (size_t)G4 * (l ? 256 : EMBD));
    cvt(eW[l][d][1], eWhhBf[l][d], (size_t)G4 * HHD);
    cvt(dW[l][d][1], dWhhBf[l][d], (size_t)G4 * HHD);
  }
  for (int d = 0; d < 2; d++) cvt(dW[1][d][0], dWihBf[d], (size_t)G4 * 256);
  cvt(p1W, p1Wbf, (size_t)512 * 512);
  cvt(p2W, p2Wbf, (size_t)512 * 512);
  cvt(prW, prWbf, (size_t)NMX * 256);

  // ---- embedding gather
  k_embed_bf<<<NB(BT * EMBD, 256), 256, 0, stream>>>(seq, emb, xbf, BT * EMBD);

  // ---- encoder layer 0 (writes cat0: [0:128]=fwd, [128:256]=bwd un-reversed)
  gemm(xbf, eWihBf[0][0], eW[0][0][2], eW[0][0][3], xpF, (int)BT, G4, EMBD);
  scan(xpF, eWhhBf[0][0], len, nullptr, nullptr, 0, cat0, 256, 0,
       hncat + 0 * HHD, cncat + 0 * HHD);
  k_rev_gather_bf<<<NB(BT * EMBD, 256), 256, 0, stream>>>(
      xbf, xrev, len, BT * EMBD, EMBD, EMBD, 0, EMBD, 0, T);
  gemm(xrev, eWihBf[0][1], eW[0][1][2], eW[0][1][3], xpF, (int)BT, G4, EMBD);
  scan(xpF, eWhhBf[0][1], len, nullptr, nullptr, 0, tmpO, HHD, 0,
       hncat + 1 * HHD, cncat + 1 * HHD);
  k_rev_gather_bf<<<NB(BT * HHD, 256), 256, 0, stream>>>(
      tmpO, cat0, len, BT * HHD, HHD, HHD, 0, 256, 128, T);

  // ---- encoder layer 1 (outputs unused; only hT/cT needed)
  gemm(cat0, eWihBf[1][0], eW[1][0][2], eW[1][0][3], xpF, (int)BT, G4, 256);
  scan(xpF, eWhhBf[1][0], len, nullptr, nullptr, 0, nullptr, 0, 0,
       hncat + 2 * HHD, cncat + 2 * HHD);
  k_rev_gather_bf<<<NB(BT * 256, 256), 256, 0, stream>>>(
      cat0, xrev, len, BT * 256, 256, 256, 0, 256, 0, T);
  gemm(xrev, eWihBf[1][1], eW[1][1][2], eW[1][1][3], xpF, (int)BT, G4, 256);
  scan(xpF, eWhhBf[1][1], len, nullptr, nullptr, 0, nullptr, 0, 0,
       hncat + 3 * HHD, cncat + 3 * HHD);

  // ---- GNNs (shared degree norm)
  k_zero<<<1, 256, 0, stream>>>(deg, Bn);
  k_deg<<<NB(E + Bn, 256), 256, 0, stream>>>(eidx, deg, E, Bn);
  k_dinv<<<1, 256, 0, stream>>>(deg, Bn);
  auto gnn = [&](const float* xin, const float* const* P, float* gout) {
    k_gemm_naive<<<NB(Bn * 16, 256), 256, 0, stream>>>(xin, P[0], nullptr, gxw, Bn, 16, 512);
    k_zero<<<NB(Bn * 16, 256), 256, 0, stream>>>(gh1, (size_t)Bn * 16);
    k_scatter<<<NB((E + Bn) * 16, 256), 256, 0, stream>>>(eidx, deg, gxw, gh1, E, Bn, 16);
    k_bias_lrelu<<<NB(Bn * 16, 256), 256, 0, stream>>>(gh1, P[1], Bn, 16);
    k_gemm_naive<<<NB(Bn * 32, 256), 256, 0, stream>>>(gh1, P[2], nullptr, gxw, Bn, 32, 16);
    k_zero<<<NB(Bn * 32, 256), 256, 0, stream>>>(gh2, (size_t)Bn * 32);
    k_scatter<<<NB((E + Bn) * 32, 256), 256, 0, stream>>>(eidx, deg, gxw, gh2, E, Bn, 32);
    k_bias_lrelu<<<NB(Bn * 32, 256), 256, 0, stream>>>(gh2, P[3], Bn, 32);
    k_gemm_naive<<<NB(Bn * 512, 256), 256, 0, stream>>>(gh2, P[4], P[5], gout, Bn, 512, 32);
  };
  gnn(hncat, g1p, goutH);
  gnn(cncat, g2p, goutC);

  // ---- projections -> decoder initial states (B,512), slice s*128 per state
  cvt(goutH, gbf, (size_t)Bn * 512);
  gemm(gbf, p1Wbf, p1b, nullptr, projH, Bn, 512, 512);
  cvt(goutC, gbf, (size_t)Bn * 512);
  gemm(gbf, p2Wbf, p2b, nullptr, projC, Bn, 512, 512);

  // ---- decoder layer 0 (mask=1, full reverse)
  k_dec_xp<<<NB(BT * G4, 256), 256, 0, stream>>>(
      seq, dW[0][0][0], dW[0][0][2], dW[0][0][3], xpF, 0, BT * G4, T);
  scan(xpF, dWhhBf[0][0], nullptr, projH + 0 * HHD, projC + 0 * HHD, 512,
       cat0, 256, 0, nullptr, nullptr);
  k_dec_xp<<<NB(BT * G4, 256), 256, 0, stream>>>(
      seq, dW[0][1][0], dW[0][1][2], dW[0][1][3], xpF, 1, BT * G4, T);
  scan(xpF, dWhhBf[0][1], nullptr, projH + 1 * HHD, projC + 1 * HHD, 512,
       tmpO, HHD, 0, nullptr, nullptr);
  k_rev_gather_bf<<<NB(BT * HHD, 256), 256, 0, stream>>>(
      tmpO, cat0, nullptr, BT * HHD, HHD, HHD, 0, 256, 128, T);

  // ---- decoder layer 1
  gemm(cat0, dWihBf[0], dW[1][0][2], dW[1][0][3], xpF, (int)BT, G4, 256);
  scan(xpF, dWhhBf[1][0], nullptr, projH + 2 * HHD, projC + 2 * HHD, 512,
       cat1, 256, 0, nullptr, nullptr);
  k_rev_gather_bf<<<NB(BT * 256, 256), 256, 0, stream>>>(
      cat0, xrev, nullptr, BT * 256, 256, 256, 0, 256, 0, T);
  gemm(xrev, dWihBf[1], dW[1][1][2], dW[1][1][3], xpF, (int)BT, G4, 256);
  scan(xpF, dWhhBf[1][1], nullptr, projH + 3 * HHD, projC + 3 * HHD, 512,
       tmpO, HHD, 0, nullptr, nullptr);
  k_rev_gather_bf<<<NB(BT * HHD, 256), 256, 0, stream>>>(
      tmpO, cat1, nullptr, BT * HHD, HHD, HHD, 0, 256, 128, T);

  // ---- prediction head: (B*T,256) @ (4096,256)^T + pred_b  -> d_out
  gemm(cat1, prWbf, prb, nullptr, out, (int)BT, NMX, 256);
}
